// MoERuntimeExperts_30167850287536
// MI455X (gfx1250) — matrix operand order repaired and checked by reading, not compiled, not run on hardware
//
#include <hip/hip_runtime.h>
#include <hip/hip_bf16.h>
#include <math.h>

// Async global->LDS path (CDNA5 GLOBAL_LOAD_ASYNC_TO_LDS_B128), guarded so the
// kernel still compiles if the toolchain lacks the builtin.
#if defined(__gfx1250__) && __has_builtin(__builtin_amdgcn_global_load_async_to_lds_b128)
#define ASYNC_A 1
#else
#define ASYNC_A 0
#endif

typedef __attribute__((ext_vector_type(16))) __bf16 v16bf;
typedef __attribute__((ext_vector_type(8)))  __bf16 v8bf;
typedef __attribute__((ext_vector_type(4)))  __bf16 v4bf;
typedef __attribute__((ext_vector_type(2)))  __bf16 v2bf;
typedef __attribute__((ext_vector_type(8)))  float  v8f;
typedef __attribute__((ext_vector_type(4)))  int    v4i;

#define TOKENS   8192
#define NEXP     8
#define DIN      1024
#define DHID     4096
#define DOUT     1024
#define BM       128
#define BN       128
#define BK       64
#define LDSS     72      // padded LDS row stride in halves (144B = 36 banks, conflict-free b128)
#define MAX_TILES 72

// ---- workspace layout (bytes) ----
#define WS_COUNTS   0          // 8 int
#define WS_CURSORS  64         // 8 int
#define WS_OFFSETS  128        // 8 int
#define WS_NUMTILES 192        // 1 int
#define WS_TILES    256        // MAX_TILES * int4
#define WS_PERM     2048       // 8192 int
#define WS_XG       65536      // bf16 [8192 x 1024]
#define WS_H        (65536 + TOKENS*DIN*2)   // bf16 [8192 x 4096]

// ------------------- routing kernels -------------------

__global__ void moe_init(int* counts, int* cursors, int* numTiles) {
    int t = threadIdx.x;
    if (t < NEXP) { counts[t] = 0; cursors[t] = 0; }
    if (t == 0) *numTiles = 0;
}

__global__ void moe_count(const long long* __restrict__ idx, int* counts) {
    int t = blockIdx.x * blockDim.x + threadIdx.x;
    if (t < TOKENS) {
        int e = (int)idx[t] & (NEXP - 1);
        atomicAdd(&counts[e], 1);
    }
}

__global__ void moe_partition(const int* __restrict__ counts, int* offsets,
                              int4* tiles, int* numTiles) {
    if (threadIdx.x != 0) return;
    int off = 0, t = 0;
    for (int e = 0; e < NEXP; ++e) {
        offsets[e] = off;
        int c = counts[e];
        for (int r = 0; r < c; r += BM) {
            int rows = c - r; if (rows > BM) rows = BM;
            tiles[t] = make_int4(e, off + r, rows, 0);
            ++t;
        }
        off += c;
    }
    *numTiles = t;
}

__global__ void moe_scatter(const long long* __restrict__ idx,
                            const int* __restrict__ offsets,
                            int* cursors, int* perm) {
    int t = blockIdx.x * blockDim.x + threadIdx.x;
    if (t < TOKENS) {
        int e = (int)idx[t] & (NEXP - 1);
        int p = atomicAdd(&cursors[e], 1);
        perm[offsets[e] + p] = t;
    }
}

// one block per permuted row: x row (f32) -> bf16 gathered matrix
__global__ void moe_gather(const float* __restrict__ x,
                           const int* __restrict__ perm,
                           __bf16* __restrict__ Xg) {
    int row = blockIdx.x;
    int token = perm[row];
    int c = threadIdx.x * 4;                    // 256 threads x float4 = 1024
    float4 v = *(const float4*)(x + (size_t)token * DIN + c);
    v4bf o;
    o[0] = (__bf16)v.x; o[1] = (__bf16)v.y; o[2] = (__bf16)v.z; o[3] = (__bf16)v.w;
    *(v4bf*)(Xg + (size_t)row * DIN + c) = o;
}

// ------------------- WMMA fragment loaders -------------------
// A-matrix 16x32 bf16 per ISA 7.12.2: lane<16 holds M=lane, K {0..7,16..23};
// lane>=16 holds M=lane-16, K {8..15,24..31}. Contiguous 8-half runs -> b128 ds loads.
__device__ __forceinline__ v16bf load_afrag(const __bf16* As, int rowBase, int lane, int ks) {
    int r  = rowBase + (lane & 15);
    int kb = (lane & 16) ? 8 : 0;
    const __bf16* p = As + r * LDSS + ks + kb;
    v8bf lo = *(const v8bf*)(p);
    v8bf hi = *(const v8bf*)(p + 16);
    return __builtin_shufflevector(lo, hi, 0,1,2,3,4,5,6,7,8,9,10,11,12,13,14,15);
}
// B-matrix 32x16 bf16: lane<16 holds N=lane, K {0..7,16..23}; lane>=16 holds N=lane-16, K {8..15,24..31}.
// Bs stored [n][k] so the 16 halves are contiguous.
__device__ __forceinline__ v16bf load_bfrag(const __bf16* Bs, int colBase, int lane, int ks) {
    int n  = colBase + (lane & 15);
    int kb = (lane & 16) ? 8 : 0;
    const __bf16* p = Bs + n * LDSS + ks + kb;
    v8bf lo = *(const v8bf*)(p);
    v8bf hi = *(const v8bf*)(p + 16);
    return __builtin_shufflevector(lo, hi, 0,1,2,3,4,5,6,7,8,9,10,11,12,13,14,15);
}

// ------------------- tiled expert GEMM (double-buffered) -------------------
template<int KDIM, bool FUSE_GELU>
__global__ __launch_bounds__(256)
void moe_gemm(const __bf16* __restrict__ A, int lda,
              const float* __restrict__ W, int ldn,
              const float* __restrict__ bias,
              void* __restrict__ Out, int ldo,
              const int* __restrict__ perm,
              const int4* __restrict__ tiles,
              const int* __restrict__ numTiles) {
    if ((int)blockIdx.y >= *numTiles) return;
    int4 tile    = tiles[blockIdx.y];
    const int expert   = tile.x;
    const int rowStart = tile.y;
    const int rows     = tile.z;
    const int n0       = blockIdx.x * BN;
    const bool fullA   = (rows == BM);      // block-uniform

    const float* We = W    + (size_t)expert * KDIM * ldn;
    const float* be = bias + (size_t)expert * ldn;

    __shared__ alignas(16) __bf16 As[2][BM * LDSS];
    __shared__ alignas(16) __bf16 Bs[2][BN * LDSS];

    const int tid  = threadIdx.x;
    const int lane = tid & 31;
    const int wid  = tid >> 5;
    const int wm   = wid >> 1;          // 0..3 -> 32-row slab
    const int wn   = wid & 1;           // 0..1 -> 64-col slab

    v8f acc[2][4] = {};

    // ---- staging helpers (per-thread: 4 A-chunks of v8bf, 4 B row-pairs of float4x2) ----
    auto stageA_async = [&](int buf, int k0) {
#if ASYNC_A
        #pragma unroll
        for (int i = 0; i < 4; ++i) {
            int c = tid + i * 256;
            int row = c >> 3, kq = (c & 7) * 8;
            const __bf16* g = A + (size_t)(rowStart + row) * lda + k0 + kq;
            const __bf16* l = &As[buf][row * LDSS + kq];
            __builtin_amdgcn_global_load_async_to_lds_b128(
                (__attribute__((address_space(1))) v4i*)(uintptr_t)g,
                (__attribute__((address_space(3))) v4i*)(uintptr_t)l, 0, 0);
        }
#else
        (void)buf; (void)k0;
#endif
    };
    auto loadA_regs = [&](int k0, v8bf aw[4]) {
        #pragma unroll
        for (int i = 0; i < 4; ++i) {
            int c = tid + i * 256;
            int row = c >> 3, kq = (c & 7) * 8;
            v8bf v = {};
            if (row < rows)
                v = *(const v8bf*)(A + (size_t)(rowStart + row) * lda + k0 + kq);
            aw[i] = v;
        }
    };
    auto storeA_lds = [&](int buf, const v8bf aw[4]) {
        #pragma unroll
        for (int i = 0; i < 4; ++i) {
            int c = tid + i * 256;
            int row = c >> 3, kq = (c & 7) * 8;
            *(v8bf*)(&As[buf][row * LDSS + kq]) = aw[i];
        }
    };
    auto loadB_regs = [&](int k0, float4 b0[4], float4 b1[4]) {
        #pragma unroll
        for (int i = 0; i < 4; ++i) {
            int c = tid + i * 256;
            int kp = c >> 5, nb = (c & 31) * 4;        // k pair 2*kp, 4 cols
            const float* g = We + (size_t)(k0 + 2 * kp) * ldn + n0 + nb;
            b0[i] = *(const float4*)(g);
            b1[i] = *(const float4*)(g + ldn);
        }
    };
    auto storeB_lds = [&](int buf, const float4 b0[4], const float4 b1[4]) {
        #pragma unroll
        for (int i = 0; i < 4; ++i) {
            int c = tid + i * 256;
            int kp = c >> 5, nb = (c & 31) * 4;
            const float* f0 = reinterpret_cast<const float*>(&b0[i]);
            const float* f1 = reinterpret_cast<const float*>(&b1[i]);
            #pragma unroll
            for (int j = 0; j < 4; ++j) {              // packed bf16x2 -> ds_store_b32
                v2bf p; p[0] = (__bf16)f0[j]; p[1] = (__bf16)f1[j];
                *(v2bf*)(&Bs[buf][(nb + j) * LDSS + 2 * kp]) = p;
            }
        }
    };
    auto async_wait = [&]() {
#if ASYNC_A
#if __has_builtin(__builtin_amdgcn_s_wait_asynccnt)
        __builtin_amdgcn_s_wait_asynccnt(0);
#else
        asm volatile("s_wait_asynccnt 0x0" ::: "memory");
#endif
#endif
    };
    auto compute = [&](int buf) {
        const __bf16* Ab = As[buf];
        const __bf16* Bb = Bs[buf];
        #pragma unroll
        for (int ks = 0; ks < BK; ks += 32) {
            v16bf a0 = load_afrag(Ab, wm * 32,      lane, ks);
            v16bf a1 = load_afrag(Ab, wm * 32 + 16, lane, ks);
            #pragma unroll
            for (int nf = 0; nf < 4; ++nf) {
                v16bf b = load_bfrag(Bb, wn * 64 + nf * 16, lane, ks);
                acc[0][nf] = __builtin_amdgcn_wmma_f32_16x16x32_bf16(
                    false, a0, false, b, (short)0, acc[0][nf], false, false);
                acc[1][nf] = __builtin_amdgcn_wmma_f32_16x16x32_bf16(
                    false, a1, false, b, (short)0, acc[1][nf], false, false);
            }
        }
    };

    // ---- prologue: stage tile 0 into buffer 0 ----
    {
        if (ASYNC_A && fullA) {
            stageA_async(0, 0);
        } else {
            v8bf aw[4];
            loadA_regs(0, aw);
            storeA_lds(0, aw);
        }
        float4 b0[4], b1[4];
        loadB_regs(0, b0, b1);
        storeB_lds(0, b0, b1);
        async_wait();
        __syncthreads();
    }

    // ---- main pipelined loop ----
    int buf = 0;
    for (int k0 = 0; k0 < KDIM; k0 += BK) {
        const bool haveNext = (k0 + BK) < KDIM;
        const int  nxt = buf ^ 1;
        float4 b0[4], b1[4];
        v8bf   aw[4];
        bool manualA = false;
        if (haveNext) {
            loadB_regs(k0 + BK, b0, b1);        // global f32 loads in flight over WMMAs
            if (ASYNC_A && fullA) {
                stageA_async(nxt, k0 + BK);     // ASYNCcnt-tracked DMA into next buffer
            } else {
                loadA_regs(k0 + BK, aw);
                manualA = true;
            }
            if (k0 + 2 * BK < KDIM)
                __builtin_prefetch((const void*)(We + (size_t)(k0 + 2 * BK) * ldn + n0 + lane * 4), 0, 0);
        }
        compute(buf);                            // 16 WMMAs on current buffer
        if (haveNext) {
            storeB_lds(nxt, b0, b1);
            if (manualA) storeA_lds(nxt, aw);
        }
        async_wait();
        __syncthreads();
        buf = nxt;
    }

    // ---- epilogue: C layout (lane<16: N=lane, M=vgpr; lane>=16: N=lane-16, M=8+vgpr) ----
    int colLane = lane & 15;
    int rowHalf = (lane & 16) ? 8 : 0;
    #pragma unroll
    for (int mf = 0; mf < 2; ++mf) {
        #pragma unroll
        for (int nf = 0; nf < 4; ++nf) {
            int col = n0 + wn * 64 + nf * 16 + colLane;
            float bcol = be[col];
            v8f a = acc[mf][nf];
            #pragma unroll
            for (int i = 0; i < 8; ++i) {
                int rTile = wm * 32 + mf * 16 + rowHalf + i;
                if (rTile < rows) {
                    float v = a[i] + bcol;
                    if constexpr (FUSE_GELU) {
                        float g = 0.5f * v * (1.0f + erff(v * 0.70710678118654752f));
                        ((__bf16*)Out)[(size_t)(rowStart + rTile) * ldo + col] = (__bf16)g;
                    } else {
                        int token = perm[rowStart + rTile];
                        ((float*)Out)[(size_t)token * ldo + col] = v;
                    }
                }
            }
        }
    }
}

// ------------------- launcher -------------------

extern "C" void kernel_launch(void* const* d_in, const int* in_sizes, int n_in,
                              void* d_out, int out_size, void* d_ws, size_t ws_size,
                              hipStream_t stream) {
    const float*     x   = (const float*)d_in[0];
    const long long* idx = (const long long*)d_in[1];   // jnp.int64 routing indices
    const float*     w1  = (const float*)d_in[2];       // [8,1024,4096]
    const float*     w2  = (const float*)d_in[3];       // [8,4096,1024]
    const float*     b1  = (const float*)d_in[4];       // [8,4096]
    const float*     b2  = (const float*)d_in[5];       // [8,1024]
    float*           out = (float*)d_out;               // [8192,1,1024] flat

    char* ws       = (char*)d_ws;
    int*  counts   = (int*)(ws + WS_COUNTS);
    int*  cursors  = (int*)(ws + WS_CURSORS);
    int*  offsets  = (int*)(ws + WS_OFFSETS);
    int*  numTiles = (int*)(ws + WS_NUMTILES);
    int4* tiles    = (int4*)(ws + WS_TILES);
    int*  perm     = (int*)(ws + WS_PERM);
    __bf16* Xg     = (__bf16*)(ws + WS_XG);
    __bf16* H      = (__bf16*)(ws + WS_H);

    moe_init     <<<1, 64, 0, stream>>>(counts, cursors, numTiles);
    moe_count    <<<TOKENS / 256, 256, 0, stream>>>(idx, counts);
    moe_partition<<<1, 32, 0, stream>>>(counts, offsets, tiles, numTiles);
    moe_scatter  <<<TOKENS / 256, 256, 0, stream>>>(idx, offsets, cursors, perm);
    moe_gather   <<<TOKENS, 256, 0, stream>>>(x, perm, Xg);

    // GEMM1: [rows,1024] x [1024,4096] + b1 -> gelu -> H (bf16)
    moe_gemm<DIN, true><<<dim3(DHID / BN, MAX_TILES), 256, 0, stream>>>(
        Xg, DIN, w1, DHID, b1, (void*)H, DHID, perm, tiles, numTiles);

    // GEMM2: [rows,4096] x [4096,1024] + b2 -> scatter f32 rows to out
    moe_gemm<DHID, false><<<dim3(DOUT / BN, MAX_TILES), 256, 0, stream>>>(
        H, DHID, w2, DOUT, b2, (void*)out, DOUT, perm, tiles, numTiles);
}
